// SequenceToPairBlock_22325240005284
// MI455X (gfx1250) — matrix operand order repaired and checked by reading, not compiled, not run on hardware
//
#include <hip/hip_runtime.h>
#include <hip/hip_bf16.h>
#include <math.h>

#define BB   2
#define LL   8192
#define CC   1536
#define PP   512
#define WWIN 16
#define HH   32
#define FF   128
#define POSC 32
#define HF   (HH * FF)      // 4096
#define NREL (2 * PP - 1)   // 1023

typedef __attribute__((ext_vector_type(16))) __bf16 v16bf;
typedef __attribute__((ext_vector_type(8)))  float  v8f;
typedef __attribute__((ext_vector_type(4)))  float  f32x4;
typedef __bf16 bf16;

union FragU { v16bf v; uint4 u[2]; };

__device__ __forceinline__ v8f vzero8() {
  v8f z = {0.f, 0.f, 0.f, 0.f, 0.f, 0.f, 0.f, 0.f};
  return z;
}

__device__ __forceinline__ v8f wmma_bf16(v16bf a, v16bf b, v8f c) {
  // D = A(16x32 bf16) * B(32x16 bf16) + C(16x16 f32)
  return __builtin_amdgcn_wmma_f32_16x16x32_bf16(false, a, false, b, (short)0, c,
                                                 false, false);
}

// ---------------- CDNA5 async global->LDS staging helpers ------------------
// Each lane copies 16B from global to LDS; tracked by ASYNCcnt (not LOADcnt).
__device__ __forceinline__ void async_copy16(const bf16* g, const bf16* lds) {
  unsigned l32 = (unsigned)(size_t)lds;   // generic LDS addr low 32 bits == LDS offset
  unsigned long long g64 = (unsigned long long)(size_t)g;
  asm volatile("global_load_async_to_lds_b128 %0, %1, off"
               :: "v"(l32), "v"(g64) : "memory");
}
__device__ __forceinline__ void wait_async0() {
  asm volatile("s_wait_asynccnt 0x0" ::: "memory");
}
__device__ __forceinline__ void wait_async1() {
  asm volatile("s_wait_asynccnt 0x1" ::: "memory");
}

// ---- A fragment: 16(M) x 32(K) bf16 tile, base = &A[m0][k0], ld in elements.
// Layout: lane<16 -> M=lane, K = half*16 + 0..15 contiguous (matches load_b).
__device__ __forceinline__ v16bf load_a(const bf16* base, int ld) {
  int l = threadIdx.x & 31;
  int row = l & 15, half = l >> 4;
  const bf16* p = base + (size_t)row * ld + half * 16;
  FragU f;
  f.u[0] = *reinterpret_cast<const uint4*>(p);        // K = half*16 + 0..7
  f.u[1] = *reinterpret_cast<const uint4*>(p + 8);    // K = half*16 + 8..15
  return f.v;
}

// ---- B fragment: 32(K) x 16(N) bf16, given B^T stored [N x K] row-major.
// Layout used here (same as load_a): lane n=l&15, K = (l>>4)*16 + 0..15 contiguous.
__device__ __forceinline__ v16bf load_b(const bf16* baseT, int ld) {
  int l = threadIdx.x & 31;
  int n = l & 15, half = l >> 4;
  const bf16* p = baseT + (size_t)n * ld + half * 16;
  FragU f;
  f.u[0] = *reinterpret_cast<const uint4*>(p);
  f.u[1] = *reinterpret_cast<const uint4*>(p + 8);
  return f.v;
}

// ---- B fragment gathered from pos_enc band rows (p = pbase + n, clamped).
__device__ __forceinline__ v16bf load_b_band(const bf16* pos, int hc_off, int pbase) {
  int l = threadIdx.x & 31;
  int n = l & 15, half = l >> 4;
  int p = pbase + n;
  p = (p < 0) ? 0 : ((p > NREL - 1) ? NREL - 1 : p);
  const bf16* q = pos + (size_t)p * HF + hc_off + half * 16;
  FragU f;
  f.u[0] = *reinterpret_cast<const uint4*>(q);
  f.u[1] = *reinterpret_cast<const uint4*>(q + 8);
  return f.v;
}

// ============================ prep kernels =================================

// dst[cols x rows] bf16 = transpose(src[rows x cols] f32); coalesced writes.
__global__ void transpose_bf16_kernel(const float* __restrict__ src,
                                      bf16* __restrict__ dst,
                                      int rows, int cols) {
  size_t i = (size_t)blockIdx.x * blockDim.x + threadIdx.x;
  size_t total = (size_t)rows * cols;
  if (i >= total) return;
  size_t n = i / rows;
  size_t k = i - n * rows;
  dst[i] = (bf16)src[k * (size_t)cols + n];
}

// AvgPool(W=16) + EMA-RMS norm; emit xn (bf16) and gelu(xn) (bf16).
__global__ void pool_norm_kernel(const float* __restrict__ x,
                                 const float* __restrict__ nscale,
                                 const float* __restrict__ nms,
                                 bf16* __restrict__ xnbf,
                                 bf16* __restrict__ xabf) {
  int bp = blockIdx.x;                 // b*PP + p
  int b = bp / PP, p = bp - b * PP;
  const float* xrow = x + ((size_t)b * LL + (size_t)p * WWIN) * CC;
  for (int c = threadIdx.x; c < CC; c += blockDim.x) {
    float s = 0.f;
    #pragma unroll
    for (int w = 0; w < WWIN; ++w) s += xrow[(size_t)w * CC + c];
    float xm = s * (1.0f / (float)WWIN);
    float xn = xm * (nscale[c] * rsqrtf(nms[c] + 1e-5f));
    xnbf[(size_t)bp * CC + c] = (bf16)xn;
    float xa = 0.5f * xn * (1.0f + erff(xn * 0.70710678118654752f));
    xabf[(size_t)bp * CC + c] = (bf16)xa;
  }
}

// pos_feat (device-recomputed, f64) -> pos_enc bf16 [NREL, H*F];
// also brq[p][h] = sum_f qrb[h,f]*pos_enc[p,h,f]; brk likewise.
__global__ void pos_kernel(const float* __restrict__ Wp, const float* __restrict__ bp,
                           const float* __restrict__ qrb, const float* __restrict__ krb,
                           bf16* __restrict__ posbf,
                           float* __restrict__ brq, float* __restrict__ brk) {
  __shared__ float feat[POSC];
  __shared__ __align__(16) float row[HF];
  int r = blockIdx.x;
  int t = threadIdx.x;
  if (t < POSC / 2) {
    // widths = t + geomspace(1, P-half+1=497, 16, endpoint=False)[t] = t + 497^(t/16)
    double w = (double)t + exp(((double)t / 16.0) * log(497.0));
    double rel = (double)(r - (PP - 1));
    float on = (w > fabs(rel)) ? 1.0f : 0.0f;
    float sg = (rel > 0.0) ? 1.0f : ((rel < 0.0) ? -1.0f : 0.0f);
    feat[t] = on;
    feat[t + POSC / 2] = on * sg;
  }
  __syncthreads();
  for (int o = t; o < HF; o += blockDim.x) {
    float acc = bp[o];
    #pragma unroll
    for (int j = 0; j < POSC; ++j) acc += feat[j] * Wp[(size_t)j * HF + o];
    row[o] = acc;
    posbf[(size_t)r * HF + o] = (bf16)acc;
  }
  __syncthreads();
  if (t < HH) {
    float a = 0.f;
    for (int f = 0; f < FF; ++f) a += qrb[t * FF + f] * row[t * FF + f];
    brq[r * HH + t] = a;
  } else if (t < 2 * HH) {
    int h = t - HH;
    float a = 0.f;
    for (int f = 0; f < FF; ++f) a += krb[h * FF + f] * row[h * FF + f];
    brk[r * HH + h] = a;
  }
}

// ===================== WMMA GEMM with async-LDS B staging ==================
// C[M,N] = A[M,K]bf16 * Bt[N,K]^T bf16.
// Workgroup (8 waves) computes 128(M) x 64(N); B^T panel (64x32 bf16 = 4KB)
// is double-buffered in LDS via GLOBAL_LOAD_ASYNC_TO_LDS_B128 (ASYNCcnt),
// giving 8x B reuse; each wave owns 16 M-rows and 4 accumulators.
template <bool BF16OUT>
__global__ void __launch_bounds__(256)
gemm_async_kernel(const bf16* __restrict__ A, const bf16* __restrict__ Bt,
                  void* __restrict__ Cout, int N, int K) {
  __shared__ __align__(16) bf16 Bs[2][64 * 32];
  const int tid = threadIdx.x;
  const int wave = tid >> 5;
  const int l = tid & 31;
  const int n0 = blockIdx.x * 64;
  const int m0 = blockIdx.y * 128 + wave * 16;

  // staging addresses: thread t copies 16B of row n = t/4, chunk = t%4
  const int sn = tid >> 2, sc = tid & 3;
  const bf16* gB = Bt + (size_t)(n0 + sn) * K + sc * 8;
  const bf16* sB[2] = { &Bs[0][sn * 32 + sc * 8], &Bs[1][sn * 32 + sc * 8] };

  // per-lane A prefetch address (row-major stream over K)
  const bf16* aRow = A + (size_t)(m0 + (l & 15)) * K + (l >> 4) * 8;

  v8f acc[4] = { vzero8(), vzero8(), vzero8(), vzero8() };

  const int S = K / 32;
  async_copy16(gB, sB[0]);                       // prologue: buffer 0, k=0
  for (int s = 0; s < S; ++s) {
    const int cur = s & 1;
    __syncthreads();                             // readers of buf[!cur] done
    if (s + 1 < S) {
      async_copy16(gB + (size_t)(s + 1) * 32, sB[cur ^ 1]);
      wait_async1();                             // retire buf[cur] (in-order)
    } else {
      wait_async0();
    }
    __syncthreads();                             // buf[cur] visible to all waves
    if (s + 2 < S)
      __builtin_prefetch(aRow + (size_t)(s + 2) * 32, 0, 1);
    v16bf a = load_a(A + (size_t)m0 * K + s * 32, K);
    const bf16* bb = &Bs[cur][0];
    #pragma unroll
    for (int t = 0; t < 4; ++t)
      acc[t] = wmma_bf16(a, load_b(bb + t * 16 * 32, 32), acc[t]);
  }

  const int half = l >> 4, nn = l & 15;
  #pragma unroll
  for (int t = 0; t < 4; ++t) {
    #pragma unroll
    for (int rI = 0; rI < 8; ++rI) {
      int m = half * 8 + rI;
      size_t idx = (size_t)(m0 + m) * N + (n0 + t * 16 + nn);
      if (BF16OUT) ((bf16*)Cout)[idx] = (bf16)acc[t][rI];
      else         ((float*)Cout)[idx] = acc[t][rI];
    }
  }
}

// ===================== attention scores (fused rel-shift) ==================
// a[b,q,k,h] = q.k + 0.25*(Qrow.pos[511+k-q] + brq + Krow.pos[511+q-k] + brk)
__global__ void __launch_bounds__(32)
score_kernel(const bf16* __restrict__ qbf, const bf16* __restrict__ kbf,
             const bf16* __restrict__ posbf,
             const float* __restrict__ brq, const float* __restrict__ brk,
             bf16* __restrict__ abf) {
  int k0 = blockIdx.x * 16;
  int q0 = blockIdx.y * 16;
  int bh = blockIdx.z;
  int b = bh / HH, h = bh - b * HH;
  int hc = h * FF;
  const bf16* Qb = qbf + (size_t)(b * PP + q0) * HF + hc;
  const bf16* Kb = kbf + (size_t)(b * PP + k0) * HF + hc;
  int pqb = (PP - 1) + (k0 - q0) - 15;   // band base for q-side (31 live rows)
  int pkb = (PP - 1) + (q0 - k0) - 15;   // band base for k-side

  v8f accqk = vzero8();
  v8f rq0 = vzero8(), rq1 = vzero8(), rk0 = vzero8(), rk1 = vzero8();
  #pragma unroll
  for (int c = 0; c < FF; c += 32) {
    v16bf aq = load_a(Qb + c, HF);
    v16bf ak = load_a(Kb + c, HF);
    v16bf bk = load_b(Kb + c, HF);                 // B^T = K tile rows
    accqk = wmma_bf16(aq, bk, accqk);              // Q x K^T
    rq0 = wmma_bf16(aq, load_b_band(posbf, hc + c, pqb),      rq0);
    rq1 = wmma_bf16(aq, load_b_band(posbf, hc + c, pqb + 16), rq1);
    rk0 = wmma_bf16(ak, load_b_band(posbf, hc + c, pkb),      rk0);
    rk1 = wmma_bf16(ak, load_b_band(posbf, hc + c, pkb + 16), rk1);
  }

  __shared__ float Rq[16][32];
  __shared__ float Rk[16][32];
  int l = threadIdx.x & 31;
  int half = l >> 4, n = l & 15;
  #pragma unroll
  for (int rI = 0; rI < 8; ++rI) {
    int m = half * 8 + rI;
    Rq[m][n] = rq0[rI]; Rq[m][16 + n] = rq1[rI];
    Rk[m][n] = rk0[rI]; Rk[m][16 + n] = rk1[rI];
  }
  __syncthreads();
  #pragma unroll
  for (int rI = 0; rI < 8; ++rI) {
    int qi = half * 8 + rI;
    int ki = n;
    int jq = ki - qi + 15;               // in [0,30]
    int jk = qi - ki + 15;               // in [0,30]
    int p1 = pqb + jq;                   // 511 + k - q
    int p2 = pkb + jk;                   // 511 + q - k
    float v = accqk[rI] +
              0.25f * (Rq[qi][jq] + brq[p1 * HH + h] +
                       Rk[ki][jk] + brk[p2 * HH + h]);
    abf[((size_t)(b * PP + q0 + qi) * PP + (k0 + ki)) * HH + h] = (bf16)v;
  }
}

// ===================== pair expansion: a x Wout + biases ===================
// Output (256MB) exceeds L2 (192MB) and is never re-read -> non-temporal stores.
__global__ void __launch_bounds__(32)
pair_kernel(const bf16* __restrict__ abf, const bf16* __restrict__ woutt,
            const float* __restrict__ bout, const float* __restrict__ yq,
            const float* __restrict__ yk, float* __restrict__ out) {
  int k0 = blockIdx.x * 16;
  int bq = blockIdx.y;                   // b*PP + q
  int b = bq / PP;
  const bf16* Arow = abf + ((size_t)bq * PP + k0) * HH;   // 16 k-rows, K=32 heads
  v16bf a = load_a(Arow, HH);

  __shared__ __align__(16) float tile[16 * FF];
  int l = threadIdx.x & 31;
  int half = l >> 4, n = l & 15;
  #pragma unroll
  for (int t = 0; t < 8; ++t) {          // N = 128 in 8 tiles of 16
    v16bf bfr = load_b(woutt + (size_t)(t * 16) * HH, HH);
    v8f acc = wmma_bf16(a, bfr, vzero8());
    #pragma unroll
    for (int rI = 0; rI < 8; ++rI) {
      int m = half * 8 + rI;
      tile[m * FF + t * 16 + n] = acc[rI];
    }
  }
  __syncthreads();

  const f32x4* t4  = reinterpret_cast<const f32x4*>(tile);
  const f32x4* bo4 = reinterpret_cast<const f32x4*>(bout);
  const f32x4* yq4 = reinterpret_cast<const f32x4*>(yq + (size_t)bq * FF);
  f32x4* o4 = reinterpret_cast<f32x4*>(out + ((size_t)bq * PP + k0) * FF);
  for (int e = l; e < 16 * FF / 4; e += 32) {
    int ki = e >> 5;                     // e / (FF/4)
    int c4 = e & 31;
    const f32x4* yk4 =
        reinterpret_cast<const f32x4*>(yk + (size_t)(b * PP + k0 + ki) * FF);
    f32x4 v = t4[e] + bo4[c4] + yq4[c4] + yk4[0 + c4];
    __builtin_nontemporal_store(v, &o4[e]);   // th:NT coalesced 8KB/tile stream
  }
}

// =============================== launch ====================================
extern "C" void kernel_launch(void* const* d_in, const int* in_sizes, int n_in,
                              void* d_out, int out_size, void* d_ws, size_t ws_size,
                              hipStream_t stream) {
  (void)in_sizes; (void)n_in; (void)out_size; (void)ws_size;
  const float* x      = (const float*)d_in[0];
  const float* nscale = (const float*)d_in[1];
  const float* nms    = (const float*)d_in[2];
  const float* Wq     = (const float*)d_in[3];
  const float* Wk     = (const float*)d_in[4];
  const float* Wp     = (const float*)d_in[5];
  const float* bpv    = (const float*)d_in[6];
  const float* qrb    = (const float*)d_in[7];
  const float* krb    = (const float*)d_in[8];
  const float* Wyq    = (const float*)d_in[9];
  const float* Wyk    = (const float*)d_in[10];
  const float* Wout   = (const float*)d_in[11];
  const float* bout   = (const float*)d_in[12];
  float* out = (float*)d_out;

  char* ws = (char*)d_ws;
  size_t off = 0;
  auto alloc = [&](size_t bytes) -> void* {
    off = (off + 255) & ~(size_t)255;
    void* p = ws + off;
    off += bytes;
    return p;
  };
  const size_t MP = (size_t)BB * PP;                 // 1024 pooled rows
  bf16* xnbf  = (bf16*)alloc(MP * CC * 2);
  bf16* xabf  = (bf16*)alloc(MP * CC * 2);
  bf16* wqt   = (bf16*)alloc((size_t)HF * CC * 2);
  bf16* wkt   = (bf16*)alloc((size_t)HF * CC * 2);
  bf16* wyqt  = (bf16*)alloc((size_t)FF * CC * 2);
  bf16* wykt  = (bf16*)alloc((size_t)FF * CC * 2);
  bf16* woutt = (bf16*)alloc((size_t)FF * HH * 2);
  bf16* posbf = (bf16*)alloc((size_t)NREL * HF * 2);
  float* brq  = (float*)alloc((size_t)NREL * HH * 4);
  float* brk  = (float*)alloc((size_t)NREL * HH * 4);
  bf16* qbf   = (bf16*)alloc(MP * HF * 2);
  bf16* kbf   = (bf16*)alloc(MP * HF * 2);
  bf16* abf   = (bf16*)alloc(MP * PP * HH * 2);
  float* yq   = (float*)alloc(MP * FF * 4);
  float* yk   = (float*)alloc(MP * FF * 4);

  auto blks = [](size_t total) { return dim3((unsigned)((total + 255) / 256)); };

  // 1) weight prep: transpose + bf16
  transpose_bf16_kernel<<<blks((size_t)CC * HF), 256, 0, stream>>>(Wq, wqt, CC, HF);
  transpose_bf16_kernel<<<blks((size_t)CC * HF), 256, 0, stream>>>(Wk, wkt, CC, HF);
  transpose_bf16_kernel<<<blks((size_t)CC * FF), 256, 0, stream>>>(Wyq, wyqt, CC, FF);
  transpose_bf16_kernel<<<blks((size_t)CC * FF), 256, 0, stream>>>(Wyk, wykt, CC, FF);
  transpose_bf16_kernel<<<blks((size_t)HH * FF), 256, 0, stream>>>(Wout, woutt, HH, FF);

  // 2) pool + norm + gelu
  pool_norm_kernel<<<(unsigned)MP, 256, 0, stream>>>(x, nscale, nms, xnbf, xabf);

  // 3) positional encodings + rel-bias tables
  pos_kernel<<<NREL, 256, 0, stream>>>(Wp, bpv, qrb, krb, posbf, brq, brk);

  // 4) projections (WMMA + async-LDS B staging); q/k bf16 out, y_q/y_k f32 out
  gemm_async_kernel<true><<<dim3(HF / 64, (unsigned)(MP / 128)), 256, 0, stream>>>(
      xnbf, wqt, qbf, HF, CC);
  gemm_async_kernel<true><<<dim3(HF / 64, (unsigned)(MP / 128)), 256, 0, stream>>>(
      xnbf, wkt, kbf, HF, CC);
  gemm_async_kernel<false><<<dim3(FF / 64, (unsigned)(MP / 128)), 256, 0, stream>>>(
      xabf, wyqt, yq, FF, CC);
  gemm_async_kernel<false><<<dim3(FF / 64, (unsigned)(MP / 128)), 256, 0, stream>>>(
      xabf, wykt, yk, FF, CC);

  // 5) fused scores (qk^T + shifted rel terms)
  score_kernel<<<dim3(PP / 16, PP / 16, BB * HH), 32, 0, stream>>>(
      qbf, kbf, posbf, brq, brk, abf);

  // 6) pair expansion + biases -> 256MB non-temporal output stream
  pair_kernel<<<dim3(PP / 16, (unsigned)MP), 32, 0, stream>>>(
      abf, woutt, bout, yq, yk, out);
}
